// WeightedSkillSAGE_44375602102647
// MI455X (gfx1250) — matrix operand order repaired and checked by reading, not compiled, not run on hardware
//
#include <hip/hip_runtime.h>
#include <hip/hip_bf16.h>
#include <math.h>

typedef __attribute__((ext_vector_type(16))) __bf16 v16bf;
typedef __attribute__((ext_vector_type(8)))  float  v8f;

#define HDIM 128

// B operand held in LDS as 16-byte chunks, XOR-swizzled for conflict-free
// banking:  chunk(col, c) holds bf16 of W[8c .. 8c+7][col], stored at
// wlds4[col*16 + (c ^ (col & 15))].  Bank = 4*(c ^ (col&15)) mod 64 -> the 16
// lanes of each half-wave sweep all 64 banks (hardware-minimum 2 passes).
union BFrag { uint4 q[2]; v16bf a; };
union BChunk { __bf16 h[8]; uint4 q; };

__device__ __forceinline__ v16bf load_b_frag(const uint4* __restrict__ wlds4,
                                             int col, int ks, int hk)
{
    const uint4* wb = wlds4 + col * 16;
    const int ln4 = col & 15;
    const int c0  = ks * 4 + hk;           // chunks c0, c0+2 = K [k0+hk*8, +8) and [k0+16+hk*8, +8)
    BFrag f;
    f.q[0] = wb[(c0    ) ^ ln4];
    f.q[1] = wb[(c0 + 2) ^ ln4];
    return f.a;
}

// Convert two contiguous float4 chunks (8 floats) into bf16 elements [e0..e0+7]
__device__ __forceinline__ void pack8(v16bf& a, int e0, const float4& c0, const float4& c1)
{
    a[e0 + 0] = (__bf16)c0.x;  a[e0 + 1] = (__bf16)c0.y;
    a[e0 + 2] = (__bf16)c0.z;  a[e0 + 3] = (__bf16)c0.w;
    a[e0 + 4] = (__bf16)c1.x;  a[e0 + 5] = (__bf16)c1.y;
    a[e0 + 6] = (__bf16)c1.z;  a[e0 + 7] = (__bf16)c1.w;
}

// Load one 16x32 bf16 A tile (ISA layout) for k-step base k0.
// Lane = M row (ln); hk selects K chunk.  Two contiguous 8-float chunks.
__device__ __forceinline__ v16bf load_a_tile(const float4* __restrict__ row4,
                                             int k0, int hk)
{
    const int f0 = (k0 + hk * 8) >> 2;
    float4 c0 = row4[f0];
    float4 c1 = row4[f0 + 1];
    float4 c2 = row4[f0 + 4];
    float4 c3 = row4[f0 + 5];
    v16bf a;
    pack8(a, 0, c0, c1);
    pack8(a, 8, c2, c3);
    return a;
}

// ---------------------------------------------------------------------------
// Y[N,128] = relu(X[N,128] @ W[128,128] + b[128])
// One block = 128 rows, 8 waves; wave w owns rows [16w,16w+16).
// ---------------------------------------------------------------------------
__global__ __launch_bounds__(256)
void gemm128_bias_relu(const float* __restrict__ X,
                       const float* __restrict__ W,
                       const float* __restrict__ b,
                       float* __restrict__ Y,
                       int N)
{
    __shared__ uint4 wlds4[HDIM * 16];     // 32 KB swizzled bf16 W
    const int tid = threadIdx.x;
    #pragma unroll
    for (int i = 0; i < 8; ++i) {          // 2048 chunks / 256 threads
        int idx = tid + i * 256;
        int col = idx >> 4, c = idx & 15;
        const float* wp = W + (size_t)(c * 8) * HDIM + col;  // W[k][col], k=8c..
        BChunk u;
        #pragma unroll
        for (int j = 0; j < 8; ++j) u.h[j] = (__bf16)wp[j * HDIM];
        wlds4[col * 16 + (c ^ (col & 15))] = u.q;
    }
    __syncthreads();

    const int wave = tid >> 5;
    const int lane = tid & 31;
    const int hk   = lane >> 4;
    const int ln   = lane & 15;
    const int rowBase = blockIdx.x * 128 + wave * 16;

    v8f acc[8];
    #pragma unroll
    for (int t = 0; t < 8; ++t)
        #pragma unroll
        for (int r = 0; r < 8; ++r) acc[t][r] = 0.0f;

    // Clamp: OOB rows only feed output rows that are never stored.
    int arow = rowBase + ln; if (arow > N - 1) arow = N - 1;
    const float4* xrow4 = (const float4*)(X + (size_t)arow * HDIM);

    #pragma unroll
    for (int ks = 0; ks < 4; ++ks) {
        v16bf a = load_a_tile(xrow4, ks * 32, hk);
        #pragma unroll
        for (int t = 0; t < 8; ++t) {
            v16bf bm = load_b_frag(wlds4, t * 16 + ln, ks, hk);
            acc[t] = __builtin_amdgcn_wmma_f32_16x16x32_bf16(
                false, a, false, bm, (short)0, acc[t], false, false);
        }
    }

    // Epilogue: wave-uniform full/partial/empty split (N multiple of 16 ->
    // the guarded path is almost never taken, but kept for generality).
    if (rowBase + 16 <= N) {
        #pragma unroll
        for (int t = 0; t < 8; ++t) {
            int col = t * 16 + ln;
            float bias = b[col];
            #pragma unroll
            for (int r = 0; r < 8; ++r) {
                float v = acc[t][r] + bias;
                Y[(size_t)(rowBase + r + hk * 8) * HDIM + col] = v > 0.0f ? v : 0.0f;
            }
        }
    } else if (rowBase < N) {
        #pragma unroll
        for (int t = 0; t < 8; ++t) {
            int col = t * 16 + ln;
            float bias = b[col];
            #pragma unroll
            for (int r = 0; r < 8; ++r) {
                int row = rowBase + r + hk * 8;
                if (row < N) {
                    float v = acc[t][r] + bias;
                    Y[(size_t)row * HDIM + col] = v > 0.0f ? v : 0.0f;
                }
            }
        }
    }
}

// ---------------------------------------------------------------------------
// out = relu(cat(Xdst@Wi + bi, Agg@Wj + bj)); out /= max(||out||2, 1e-12);
// Out[row] (+)= out.  Column tiles 0-3 use Xdst as A, tiles 4-7 use Agg.
// LDS holds bf16 [Wi | Wj] in the swizzled chunk layout.
// ---------------------------------------------------------------------------
template <int ACC>
__global__ __launch_bounds__(256)
void sage_cat_norm(const float* __restrict__ Xdst,
                   const float* __restrict__ Agg,
                   const float* __restrict__ Wi, const float* __restrict__ bi,
                   const float* __restrict__ Wj, const float* __restrict__ bj,
                   float* __restrict__ Out,
                   int N)
{
    __shared__ uint4 wlds4[HDIM * 16];
    const int tid = threadIdx.x;
    #pragma unroll
    for (int i = 0; i < 8; ++i) {
        int idx = tid + i * 256;
        int col = idx >> 4, c = idx & 15;
        const float* wp = (col < 64) ? (Wi + (size_t)(c * 8) * 64 + col)
                                     : (Wj + (size_t)(c * 8) * 64 + (col - 64));
        BChunk u;
        #pragma unroll
        for (int j = 0; j < 8; ++j) u.h[j] = (__bf16)wp[j * 64];
        wlds4[col * 16 + (c ^ (col & 15))] = u.q;
    }
    __syncthreads();

    const int wave = tid >> 5;
    const int lane = tid & 31;
    const int hk   = lane >> 4;
    const int ln   = lane & 15;
    const int rowBase = blockIdx.x * 128 + wave * 16;

    v8f acc[8];
    #pragma unroll
    for (int t = 0; t < 8; ++t)
        #pragma unroll
        for (int r = 0; r < 8; ++r) acc[t][r] = 0.0f;

    int arow = rowBase + ln; if (arow > N - 1) arow = N - 1;
    const float4* drow4 = (const float4*)(Xdst + (size_t)arow * HDIM);
    const float4* grow4 = (const float4*)(Agg  + (size_t)arow * HDIM);

    #pragma unroll
    for (int ks = 0; ks < 4; ++ks) {
        v16bf ad = load_a_tile(drow4, ks * 32, hk);
        v16bf ag = load_a_tile(grow4, ks * 32, hk);
        #pragma unroll
        for (int t = 0; t < 8; ++t) {
            v16bf bm = load_b_frag(wlds4, t * 16 + ln, ks, hk);
            v16bf a = (t < 4) ? ad : ag;
            acc[t] = __builtin_amdgcn_wmma_f32_16x16x32_bf16(
                false, a, false, bm, (short)0, acc[t], false, false);
        }
    }

    // bias + relu in place
    #pragma unroll
    for (int t = 0; t < 8; ++t) {
        int col = t * 16 + ln;
        float bias = (col < 64) ? bi[col] : bj[col - 64];
        #pragma unroll
        for (int r = 0; r < 8; ++r) {
            float v = acc[t][r] + bias;
            acc[t][r] = v > 0.0f ? v : 0.0f;
        }
    }
    // per-row inverse L2 norm: reduce over tiles in-register, then across the
    // 16 lanes of this half (xor masks 1..8 stay inside a half on wave32)
    float inv[8];
    #pragma unroll
    for (int r = 0; r < 8; ++r) {
        float s = 0.0f;
        #pragma unroll
        for (int t = 0; t < 8; ++t) s += acc[t][r] * acc[t][r];
        #pragma unroll
        for (int m = 1; m < 16; m <<= 1) s += __shfl_xor(s, m, 32);
        inv[r] = 1.0f / fmaxf(sqrtf(s), 1e-12f);
    }

    if (rowBase + 16 <= N) {
        #pragma unroll
        for (int t = 0; t < 8; ++t) {
            int col = t * 16 + ln;
            #pragma unroll
            for (int r = 0; r < 8; ++r) {
                size_t off = (size_t)(rowBase + r + hk * 8) * HDIM + col;
                float v = acc[t][r] * inv[r];
                Out[off] = ACC ? (Out[off] + v) : v;
            }
        }
    } else if (rowBase < N) {
        #pragma unroll
        for (int t = 0; t < 8; ++t) {
            int col = t * 16 + ln;
            #pragma unroll
            for (int r = 0; r < 8; ++r) {
                int row = rowBase + r + hk * 8;
                if (row < N) {
                    size_t off = (size_t)row * HDIM + col;
                    float v = acc[t][r] * inv[r];
                    Out[off] = ACC ? (Out[off] + v) : v;
                }
            }
        }
    }
}

// ---------------------------------------------------------------------------
// agg[dst[e]][c] = max(agg, X0[src[e]][c] * w[e]); messages are >= 0 (relu *
// uniform[0,1)), so integer atomicMax on float bits is exact & deterministic.
// Each thread handles 4 channels of one edge (float4 load, 4 atomics).
// ---------------------------------------------------------------------------
__global__ __launch_bounds__(256)
void edge_scatter_max(const float* __restrict__ X0,
                      const int* __restrict__ src,
                      const int* __restrict__ dst,
                      const float* __restrict__ w,
                      int* __restrict__ agg,
                      int E_)
{
    long long idx = (long long)blockIdx.x * blockDim.x + threadIdx.x;
    long long total = (long long)E_ * 32;          // 128 ch / 4 per thread
    if (idx >= total) return;
    int e  = (int)(idx >> 5);
    int c4 = (int)(idx & 31) * 4;
    int s = src[e], d = dst[e];
    float ew = w[e];
    float4 m = *(const float4*)(X0 + (size_t)s * HDIM + c4);
    int* ap = agg + (size_t)d * HDIM + c4;
    atomicMax(ap + 0, __float_as_int(m.x * ew));
    atomicMax(ap + 1, __float_as_int(m.y * ew));
    atomicMax(ap + 2, __float_as_int(m.z * ew));
    atomicMax(ap + 3, __float_as_int(m.w * ew));
}

__global__ __launch_bounds__(256)
void fill_zero_f32(float* __restrict__ p, long long n)
{
    long long i = (long long)blockIdx.x * blockDim.x + threadIdx.x;
    long long stride = (long long)gridDim.x * blockDim.x;
    for (; i < n; i += stride) p[i] = 0.0f;
}

// ---------------------------------------------------------------------------
extern "C" void kernel_launch(void* const* d_in, const int* in_sizes, int n_in,
                              void* d_out, int out_size, void* d_ws, size_t ws_size,
                              hipStream_t stream)
{
    (void)in_sizes; (void)n_in; (void)out_size; (void)ws_size;
    const int NSn = 100000, NJn = 100000, En = 600000;

    const float* x_skill = (const float*)d_in[0];
    const float* x_job   = (const float*)d_in[1];
    const int*   e_req   = (const int*)  d_in[2];
    const float* w_req   = (const float*)d_in[3];
    const int*   e_ss    = (const int*)  d_in[4];
    const float* w_ss    = (const float*)d_in[5];
    const int*   e_jj    = (const int*)  d_in[6];
    const float* w_jj    = (const float*)d_in[7];
    const float* W_in_s  = (const float*)d_in[8];
    const float* b_in_s  = (const float*)d_in[9];
    const float* W_in_j  = (const float*)d_in[10];
    const float* b_in_j  = (const float*)d_in[11];
    const float* W_mp    = (const float*)d_in[12];
    const float* b_mp    = (const float*)d_in[13];
    const float* W_i     = (const float*)d_in[14];
    const float* b_i     = (const float*)d_in[15];
    const float* W_j     = (const float*)d_in[16];
    const float* b_j     = (const float*)d_in[17];
    const float* W_out_s = (const float*)d_in[18];
    const float* b_out_s = (const float*)d_in[19];
    const float* W_out_j = (const float*)d_in[20];
    const float* b_out_j = (const float*)d_in[21];

    const size_t NF = (size_t)100000 * HDIM;
    float* xs   = (float*)d_ws;
    float* xj   = xs   + NF;
    float* accS = xj   + NF;
    float* accJ = accS + NF;
    float* X0   = accJ + NF;
    float* Agg  = X0   + NF;

    const dim3 blk(256);
    const int  gN = (100000 + 127) / 128;
    const long long edgeTotal = (long long)En * 32;
    const int  edgeBlocks = (int)((edgeTotal + 255) / 256);

    // input projections
    gemm128_bias_relu<<<gN, blk, 0, stream>>>(x_skill, W_in_s, b_in_s, xs, NSn);
    gemm128_bias_relu<<<gN, blk, 0, stream>>>(x_job,   W_in_j, b_in_j, xj, NJn);

    for (int l = 0; l < 2; ++l) {
        const float* mpW[4]; const float* mpB[4];
        const float* iW[4];  const float* iB[4];
        const float* jW[4];  const float* jB[4];
        for (int c = 0; c < 4; ++c) {
            size_t o = (size_t)(l * 4 + c);
            mpW[c] = W_mp + o * 128 * 128;  mpB[c] = b_mp + o * 128;
            iW[c]  = W_i  + o * 128 * 64;   iB[c]  = b_i  + o * 64;
            jW[c]  = W_j  + o * 128 * 64;   jB[c]  = b_j  + o * 64;
        }

        // ---- skill destination (accS) ----
        // conv0: Job -> Skill (REQUIRES)
        gemm128_bias_relu<<<gN, blk, 0, stream>>>(xj, mpW[0], mpB[0], X0, NJn);
        fill_zero_f32<<<2048, blk, 0, stream>>>(Agg, (long long)NSn * HDIM);
        edge_scatter_max<<<edgeBlocks, blk, 0, stream>>>(X0, e_req, e_req + En, w_req, (int*)Agg, En);
        sage_cat_norm<0><<<gN, blk, 0, stream>>>(xs, Agg, iW[0], iB[0], jW[0], jB[0], accS, NSn);
        // conv2: Skill<->Skill (both directions share the max-pool projection)
        gemm128_bias_relu<<<gN, blk, 0, stream>>>(xs, mpW[2], mpB[2], X0, NSn);
        fill_zero_f32<<<2048, blk, 0, stream>>>(Agg, (long long)NSn * HDIM);
        edge_scatter_max<<<edgeBlocks, blk, 0, stream>>>(X0, e_ss, e_ss + En, w_ss, (int*)Agg, En);
        sage_cat_norm<1><<<gN, blk, 0, stream>>>(xs, Agg, iW[2], iB[2], jW[2], jB[2], accS, NSn);
        fill_zero_f32<<<2048, blk, 0, stream>>>(Agg, (long long)NSn * HDIM);
        edge_scatter_max<<<edgeBlocks, blk, 0, stream>>>(X0, e_ss + En, e_ss, w_ss, (int*)Agg, En);
        sage_cat_norm<1><<<gN, blk, 0, stream>>>(xs, Agg, iW[2], iB[2], jW[2], jB[2], accS, NSn);

        // ---- job destination (accJ) ----
        // conv1: Skill -> Job (rev_REQUIRES)
        gemm128_bias_relu<<<gN, blk, 0, stream>>>(xs, mpW[1], mpB[1], X0, NSn);
        fill_zero_f32<<<2048, blk, 0, stream>>>(Agg, (long long)NJn * HDIM);
        edge_scatter_max<<<edgeBlocks, blk, 0, stream>>>(X0, e_req + En, e_req, w_req, (int*)Agg, En);
        sage_cat_norm<0><<<gN, blk, 0, stream>>>(xj, Agg, iW[1], iB[1], jW[1], jB[1], accJ, NJn);
        // conv3: Job<->Job
        gemm128_bias_relu<<<gN, blk, 0, stream>>>(xj, mpW[3], mpB[3], X0, NJn);
        fill_zero_f32<<<2048, blk, 0, stream>>>(Agg, (long long)NJn * HDIM);
        edge_scatter_max<<<edgeBlocks, blk, 0, stream>>>(X0, e_jj, e_jj + En, w_jj, (int*)Agg, En);
        sage_cat_norm<1><<<gN, blk, 0, stream>>>(xj, Agg, iW[3], iB[3], jW[3], jB[3], accJ, NJn);
        fill_zero_f32<<<2048, blk, 0, stream>>>(Agg, (long long)NJn * HDIM);
        edge_scatter_max<<<edgeBlocks, blk, 0, stream>>>(X0, e_jj + En, e_jj, w_jj, (int*)Agg, En);
        sage_cat_norm<1><<<gN, blk, 0, stream>>>(xj, Agg, iW[3], iB[3], jW[3], jB[3], accJ, NJn);

        // swap current features with accumulators
        float* t;
        t = xs; xs = accS; accS = t;
        t = xj; xj = accJ; accJ = t;
    }

    // output projections (relu, no normalize); d_out = [out_s | out_j]
    float* out = (float*)d_out;
    gemm128_bias_relu<<<gN, blk, 0, stream>>>(xs, W_out_s, b_out_s, out,      NSn);
    gemm128_bias_relu<<<gN, blk, 0, stream>>>(xj, W_out_j, b_out_j, out + NF, NJn);
}
